// PNAConvSC_38319698215462
// MI455X (gfx1250) — matrix operand order repaired and checked by reading, not compiled, not run on hardware
//
#include <hip/hip_runtime.h>
#include <stdint.h>

// ---------------------------------------------------------------------------
// PNAConvSC for MI455X (gfx1250): bf16 WMMA GEMMs, async-LDS-staged weights,
// L2-resident scatter with native float atomics (global_atomic_{max,min}_num_f32)
// ---------------------------------------------------------------------------

typedef __attribute__((ext_vector_type(16))) __bf16       bf16x16;
typedef __attribute__((ext_vector_type(8)))  float        f32x8;
typedef __attribute__((ext_vector_type(4)))  unsigned int u32x4;

union Frag {            // one WMMA 16-bit A or B operand: 16 bf16 per lane (8 VGPRs)
  bf16x16 v;
  u32x4   q[2];
};

static __device__ __forceinline__ unsigned short f2bf(float f) {
  // round-to-nearest-even float32 -> bf16
  unsigned int u = __float_as_uint(f);
  unsigned int r = u + 0x7FFFu + ((u >> 16) & 1u);
  return (unsigned short)(r >> 16);
}

static __device__ __forceinline__ Frag load_frag(const unsigned short* p) {
  Frag f;
  f.q[0] = *(const u32x4*)(p);      // 16B aligned
  f.q[1] = *(const u32x4*)(p + 16); // k+16 half
  return f;
}

static __device__ __forceinline__ f32x8 wmma_bf16(const Frag& a, const Frag& b, f32x8 c) {
  return __builtin_amdgcn_wmma_f32_16x16x32_bf16(false, a.v, false, b.v, (short)0, c,
                                                 false, false);
}

// Native CDNA5 float atomics (ISA 15.18.3 op 81/82) — no sign branches, no EXEC games.
static __device__ __forceinline__ void atomicMaxF(float* a, float v) {
  asm volatile("global_atomic_max_num_f32 %0, %1, off scope:SCOPE_DEV"
               :: "v"((unsigned long long)(uintptr_t)a), "v"(v) : "memory");
}
static __device__ __forceinline__ void atomicMinF(float* a, float v) {
  asm volatile("global_atomic_min_num_f32 %0, %1, off scope:SCOPE_DEV"
               :: "v"((unsigned long long)(uintptr_t)a), "v"(v) : "memory");
}

// CDNA5 async copy: one instruction per 16B, global -> LDS, tracked by ASYNCcnt.
// (ISA §10.7 GLOBAL_LOAD_ASYNC_TO_LDS_B128, op 98). LDS address = low 32 bits of
// the generic pointer (LDS aperture puts the byte offset in addr[31:0]).
static __device__ __forceinline__ void async_copy_b128(void* lds_dst, const void* gsrc) {
  asm volatile("global_load_async_to_lds_b128 %0, %1, off"
               :: "v"((unsigned)(uintptr_t)lds_dst),
                  "v"((unsigned long long)(uintptr_t)gsrc)
               : "memory");
}
static __device__ __forceinline__ void wait_async_all() {
  asm volatile("s_wait_asynccnt 0x0" ::: "memory");
}

// ---------------------------------------------------------------------------
// K0: init aggregation buffers, convert x -> bf16 (into feat cols 0..127),
//     transpose+convert weights to Wt[n][k] bf16.
// ---------------------------------------------------------------------------
__global__ void pna_init_kernel(const float* __restrict__ x,
                                const float* __restrict__ wpre,
                                const float* __restrict__ wpost,
                                const float* __restrict__ wlin,
                                float* __restrict__ cnt,
                                float* __restrict__ asum, float* __restrict__ asq,
                                float* __restrict__ amin, float* __restrict__ amax,
                                unsigned short* __restrict__ feat,
                                unsigned short* __restrict__ wtpre,
                                unsigned short* __restrict__ wtpost,
                                unsigned short* __restrict__ wtlin, int N) {
  const float INF = __int_as_float(0x7f800000);
  const int total = N * 128;
  const int stride = gridDim.x * blockDim.x;
  for (int i = blockIdx.x * blockDim.x + threadIdx.x; i < total; i += stride) {
    asum[i] = 0.f; asq[i] = 0.f; amin[i] = INF; amax[i] = -INF;
    feat[(size_t)(i >> 7) * 640 + (i & 127)] = f2bf(x[i]);
    if (i < N) cnt[i] = 0.f;
    if (i < 256 * 128)  { int r = i >> 7, c = i & 127; wtpre[c * 256 + r]            = f2bf(wpre[i]);  }
    if (i < 1664 * 128) { int r = i / 128, c = i % 128; wtpost[(size_t)c * 1664 + r] = f2bf(wpost[i]); }
    if (i < 128 * 128)  { int r = i >> 7, c = i & 127; wtlin[c * 128 + r]            = f2bf(wlin[i]);  }
  }
}

// ---------------------------------------------------------------------------
// K1: in-degree of dst nodes (float counts, L2-resident atomics)
// ---------------------------------------------------------------------------
__global__ void pna_degree_kernel(const long long* __restrict__ ei,
                                  float* __restrict__ cnt, int E) {
  const int stride = gridDim.x * blockDim.x;
  for (int e = blockIdx.x * blockDim.x + threadIdx.x; e < E; e += stride)
    atomicAdd(cnt + (int)ei[(size_t)E + e], 1.0f);
}

// ---------------------------------------------------------------------------
// K2: edge MLP (16 edges / wave-tile, bf16 WMMA) fused with scatter
//     sum / sumsq / min / max. W_pre (64 KB bf16, transposed) async-staged in
//     LDS once per block; each wave grid-strides over many tiles.
// ---------------------------------------------------------------------------
__global__ __launch_bounds__(256) void pna_edge_kernel(
    const long long* __restrict__ ei, const unsigned short* __restrict__ feat,
    const unsigned short* __restrict__ wtpre, const float* __restrict__ bpre,
    float* __restrict__ asum, float* __restrict__ asq,
    float* __restrict__ amin, float* __restrict__ amax, int E, int ntiles) {
  __shared__ unsigned short lw[256 * 128];        // Wt_pre: [n=128][k=256] bf16, 64 KB

  // async cooperative fill: 4096 x b128, 16 per thread
#pragma unroll
  for (int i = 0; i < 16; ++i) {
    const int idx = i * 256 + threadIdx.x;        // b128 index
    async_copy_b128(lw + idx * 8, wtpre + idx * 8);
  }

  const int lane  = threadIdx.x & 31;
  const int wave  = threadIdx.x >> 5;
  const int row   = lane & 15;                // A row / B column
  const int khalf = (lane >> 4) << 3;         // A: lanes 16-31 take K+8 half
  const int kboff = (lane < 16) ? 0 : 8;      // B: lanes 16-31 take K+8 half
  const int mofs  = (lane >> 4) << 3;         // C/D: lanes 16-31 hold rows M+8

  float bias8[8];
#pragma unroll
  for (int nt = 0; nt < 8; ++nt) bias8[nt] = bpre[nt * 16 + row];

  wait_async_all();                           // own async loads landed in LDS
  __syncthreads();                            // everyone's loads visible

  const int wstride = gridDim.x * 8;
  for (int tile = blockIdx.x * 8 + wave; tile < ntiles; tile += wstride) {
    const int e    = tile * 16 + row;
    const int srcN = (int)ei[e];
    const int dstN = (int)ei[(size_t)E + e];
    const unsigned short* rowDst = feat + (size_t)dstN * 640;  // x in cols 0..127
    const unsigned short* rowSrc = feat + (size_t)srcN * 640;

    // A tile: 16 edges x 256 concat features = 8 K-slabs, preloaded once
    Frag afr[8];
#pragma unroll
    for (int ks = 0; ks < 8; ++ks) {
      const unsigned short* ar = (ks < 4) ? (rowDst + ks * 32) : (rowSrc + (ks - 4) * 32);
      afr[ks].q[0] = *(const u32x4*)(ar + khalf);
      afr[ks].q[1] = *(const u32x4*)(ar + khalf + 16);
    }

    // scatter element offsets: invariant across output n-tiles, hoisted
    size_t ad8[8];
#pragma unroll
    for (int i = 0; i < 8; ++i) {
      const int dn = __shfl(dstN, i + mofs);  // dst of row (i+mofs) lives there
      ad8[i] = (size_t)dn * 128 + row;
    }

#pragma unroll
    for (int nt = 0; nt < 8; ++nt) {
      f32x8 acc = {0.f, 0.f, 0.f, 0.f, 0.f, 0.f, 0.f, 0.f};
      const unsigned short* wrow = lw + (nt * 16 + row) * 256 + kboff;
#pragma unroll
      for (int ks = 0; ks < 8; ++ks) {
        Frag b = load_frag(wrow + ks * 32);   // ds_load_b128 x2
        acc = wmma_bf16(afr[ks], b, acc);
      }
#pragma unroll
      for (int i = 0; i < 8; ++i) {
        const float  h  = acc[i] + bias8[nt];
        const size_t ad = ad8[i] + nt * 16;   // nt*16*4B folds into VMEM offset
        atomicAdd(asum + ad, h);
        atomicAdd(asq + ad, h * h);
        atomicMaxF(amax + ad, h);
        atomicMinF(amin + ad, h);
      }
    }
  }
}

// ---------------------------------------------------------------------------
// K3: finalize per-node aggregates -> feat cols 128..639 (mean|min|max|std),
//     plus amp / 1/amp scalers.
// ---------------------------------------------------------------------------
__global__ void pna_finalize_kernel(const float* __restrict__ cnt,
                                    const float* __restrict__ asum,
                                    const float* __restrict__ asq,
                                    const float* __restrict__ amin,
                                    const float* __restrict__ amax,
                                    unsigned short* __restrict__ feat,
                                    float* __restrict__ ampv,
                                    float* __restrict__ iampv, int N) {
  const int total = N * 128;
  const int stride = gridDim.x * blockDim.x;
  for (int i = blockIdx.x * blockDim.x + threadIdx.x; i < total; i += stride) {
    const int n = i >> 7, f = i & 127;
    const float c   = cnt[n];
    const float deg = fmaxf(c, 1.0f);
    const float mean = asum[i] / deg;
    const float var  = asq[i] / deg - mean * mean;
    const float stdv = sqrtf(fmaxf(var, 0.f) + 1e-5f);
    const bool  has  = c > 0.f;
    unsigned short* fr = feat + (size_t)n * 640;
    fr[128 + f] = f2bf(mean);
    fr[256 + f] = f2bf(has ? amin[i] : 0.f);
    fr[384 + f] = f2bf(has ? amax[i] : 0.f);
    fr[512 + f] = f2bf(stdv);
    if (f == 0) {
      const float a = logf(deg + 1.0f) * (1.0f / 2.8332133440562162f); // 1/log(17)
      ampv[n] = a; iampv[n] = 1.0f / a;
    }
  }
}

// ---------------------------------------------------------------------------
// K4: post MLP. (aggr*amp)@W == amp*(aggr@W): 3 accumulators over the 512-wide
//     aggregate + 1 over x. Per-nt 52 KB W_post slab async-staged in LDS and
//     shared by all 8 waves (barrier-bracketed; uniform flow, no early return).
// ---------------------------------------------------------------------------
__global__ __launch_bounds__(256) void pna_post_kernel(
    const unsigned short* __restrict__ feat, const unsigned short* __restrict__ wtpost,
    const float* __restrict__ bpost, const float* __restrict__ ampv,
    const float* __restrict__ iampv, unsigned short* __restrict__ postb, int ntiles) {
  __shared__ unsigned short lwp[16 * 1664];   // one n-tile slab of Wt_post, 52 KB

  const int lane  = threadIdx.x & 31;
  const int tile  = blockIdx.x * 8 + (threadIdx.x >> 5);
  const bool valid = (tile < ntiles);         // wave-uniform; EXEC full inside
  const int row   = lane & 15;
  const int khalf = (lane >> 4) << 3;
  const int kboff = (lane < 16) ? 0 : 8;
  const int mofs  = (lane >> 4) << 3;
  const int node  = tile * 16 + row;
  const unsigned short* frow = feat + (size_t)(valid ? node : 0) * 640;

  float av8[8], iv8[8], bb8[8];
#pragma unroll
  for (int i = 0; i < 8; ++i) {
    const int nm = valid ? (tile * 16 + i + mofs) : 0;
    av8[i] = ampv[nm]; iv8[i] = iampv[nm];
  }
#pragma unroll
  for (int nt = 0; nt < 8; ++nt) bb8[nt] = bpost[nt * 16 + row];

  for (int nt = 0; nt < 8; ++nt) {
    __syncthreads();                          // previous slab fully consumed
    {   // async slab load: 16 rows x 1664 bf16 = 3328 x b128, 13 per thread
      const unsigned short* src = wtpost + (size_t)nt * 16 * 1664;
#pragma unroll
      for (int i = 0; i < 13; ++i) {
        const int idx = i * 256 + threadIdx.x;
        async_copy_b128(lwp + idx * 8, src + idx * 8);
      }
      wait_async_all();
    }
    __syncthreads();                          // slab visible to all waves
    if (valid) {
      f32x8 aid = {0.f,0.f,0.f,0.f,0.f,0.f,0.f,0.f};
      f32x8 aam = aid, aat = aid;
      const unsigned short* wr = lwp + row * 1664 + kboff;
      // x region: W_post rows 0..127
#pragma unroll
      for (int ks = 0; ks < 4; ++ks) {
        Frag a;
        a.q[0] = *(const u32x4*)(frow + ks * 32 + khalf);
        a.q[1] = *(const u32x4*)(frow + ks * 32 + khalf + 16);
        Frag b = load_frag(wr + ks * 32);
        aid = wmma_bf16(a, b, aid);
      }
      // aggregate region: identity (128..639), amp (640..1151), att (1152..1663)
#pragma unroll 4
      for (int ks = 0; ks < 16; ++ks) {
        const unsigned short* ar = frow + 128 + ks * 32;
        Frag a;
        a.q[0] = *(const u32x4*)(ar + khalf);
        a.q[1] = *(const u32x4*)(ar + khalf + 16);
        Frag b0 = load_frag(wr + 128  + ks * 32);
        Frag b1 = load_frag(wr + 640  + ks * 32);
        Frag b2 = load_frag(wr + 1152 + ks * 32);
        aid = wmma_bf16(a, b0, aid);
        aam = wmma_bf16(a, b1, aam);
        aat = wmma_bf16(a, b2, aat);
      }
      const int col = nt * 16 + row;
#pragma unroll
      for (int i = 0; i < 8; ++i) {
        const int nm = tile * 16 + i + mofs;
        const float o = aid[i] + av8[i] * aam[i] + iv8[i] * aat[i] + bb8[nt];
        postb[(size_t)nm * 128 + col] = f2bf(o);
      }
    }
  }
}

// ---------------------------------------------------------------------------
// K5: final linear + bias + residual. Full Wt_lin (32 KB) async-staged in LDS.
// ---------------------------------------------------------------------------
__global__ __launch_bounds__(256) void pna_lin_kernel(
    const unsigned short* __restrict__ postb, const unsigned short* __restrict__ wtlin,
    const float* __restrict__ blin, const float* __restrict__ x,
    float* __restrict__ out, int ntiles) {
  __shared__ unsigned short lwl[128 * 128];   // Wt_lin: [n=128][k=128] bf16, 32 KB
#pragma unroll
  for (int i = 0; i < 8; ++i) {
    const int idx = i * 256 + threadIdx.x;
    async_copy_b128(lwl + idx * 8, wtlin + idx * 8);
  }

  const int lane  = threadIdx.x & 31;
  const int tile  = blockIdx.x * 8 + (threadIdx.x >> 5);
  const bool valid = (tile < ntiles);
  const int row   = lane & 15;
  const int khalf = (lane >> 4) << 3;
  const int kboff = (lane < 16) ? 0 : 8;
  const int mofs  = (lane >> 4) << 3;
  const int node  = tile * 16 + row;

  wait_async_all();
  __syncthreads();

  if (valid) {
    const unsigned short* frow = postb + (size_t)node * 128;
    Frag afr[4];
#pragma unroll
    for (int ks = 0; ks < 4; ++ks) {
      afr[ks].q[0] = *(const u32x4*)(frow + ks * 32 + khalf);
      afr[ks].q[1] = *(const u32x4*)(frow + ks * 32 + khalf + 16);
    }

#pragma unroll
    for (int nt = 0; nt < 8; ++nt) {
      f32x8 acc = {0.f,0.f,0.f,0.f,0.f,0.f,0.f,0.f};
      const unsigned short* wr = lwl + (nt * 16 + row) * 128 + kboff;
#pragma unroll
      for (int ks = 0; ks < 4; ++ks) {
        Frag b = load_frag(wr + ks * 32);
        acc = wmma_bf16(afr[ks], b, acc);
      }
      const int col = nt * 16 + row;
      const float bb = blin[col];
#pragma unroll
      for (int i = 0; i < 8; ++i) {
        const int nm = tile * 16 + i + mofs;
        out[(size_t)nm * 128 + col] = acc[i] + bb + x[(size_t)nm * 128 + col];
      }
    }
  }
}

// ---------------------------------------------------------------------------
// Host launcher
// ---------------------------------------------------------------------------
extern "C" void kernel_launch(void* const* d_in, const int* in_sizes, int n_in,
                              void* d_out, int out_size, void* d_ws, size_t ws_size,
                              hipStream_t stream) {
  const int N = in_sizes[0] / 128;   // 50000
  const int E = in_sizes[1] / 2;     // 800000

  const float*     x     = (const float*)d_in[0];
  const long long* ei    = (const long long*)d_in[1];
  const float*     wpre  = (const float*)d_in[2];
  const float*     bpre  = (const float*)d_in[3];
  const float*     wpost = (const float*)d_in[4];
  const float*     bpost = (const float*)d_in[5];
  const float*     wlin  = (const float*)d_in[6];
  const float*     blin  = (const float*)d_in[7];
  float*           out   = (float*)d_out;

  char* ws = (char*)d_ws;
  size_t off = 0;
  auto take = [&](size_t bytes) -> char* {
    char* p = ws + off;
    off = (off + bytes + 255) & ~(size_t)255;
    return p;
  };
  float* cnt   = (float*)take((size_t)N * 4);
  float* asum  = (float*)take((size_t)N * 128 * 4);
  float* asq   = (float*)take((size_t)N * 128 * 4);
  float* amin  = (float*)take((size_t)N * 128 * 4);
  float* amax  = (float*)take((size_t)N * 128 * 4);
  float* ampv  = (float*)take((size_t)N * 4);
  float* iampv = (float*)take((size_t)N * 4);
  unsigned short* feat   = (unsigned short*)take((size_t)N * 640 * 2);
  unsigned short* wtpre  = (unsigned short*)take((size_t)256 * 128 * 2);
  unsigned short* wtpost = (unsigned short*)take((size_t)1664 * 128 * 2);
  unsigned short* wtlin  = (unsigned short*)take((size_t)128 * 128 * 2);
  unsigned short* postb  = (unsigned short*)take((size_t)N * 128 * 2);
  (void)ws_size; (void)n_in; (void)out_size;

  // K0: init + convert + weight transpose
  pna_init_kernel<<<4096, 256, 0, stream>>>(x, wpre, wpost, wlin, cnt, asum, asq,
                                            amin, amax, feat, wtpre, wtpost, wtlin, N);
  // K1: degrees
  pna_degree_kernel<<<2048, 256, 0, stream>>>(ei, cnt, E);
  // K2: edge MLP + scatter; async-LDS-staged W_pre, waves grid-stride over tiles
  const int tilesE = (E + 15) / 16;
  pna_edge_kernel<<<1024, 256, 0, stream>>>(ei, feat, wtpre, bpre,
                                            asum, asq, amin, amax, E, tilesE);
  // K3: finalize aggregates + scalers
  pna_finalize_kernel<<<4096, 256, 0, stream>>>(cnt, asum, asq, amin, amax,
                                                feat, ampv, iampv, N);
  // K4: post MLP (one 16-node tile per wave; per-nt async LDS weight slabs)
  const int tilesN  = (N + 15) / 16;
  const int blocksN = (tilesN + 7) / 8;
  pna_post_kernel<<<blocksN, 256, 0, stream>>>(feat, wtpost, bpost, ampv, iampv,
                                               postb, tilesN);
  // K5: final linear + residual (full Wt_lin async-staged in LDS)
  pna_lin_kernel<<<blocksN, 256, 0, stream>>>(postb, wtlin, blin, x, out, tilesN);
}
